// CausalSelfAttention_43533788513028
// MI455X (gfx1250) — matrix operand order repaired and checked by reading, not compiled, not run on hardware
//
#include <hip/hip_runtime.h>

// ---------------------------------------------------------------------------
// CDNA5 (gfx1250, wave32) causal self-attention forward.
// All matmuls via v_wmma_f32_16x16x32_bf16 (bf16 operands, f32 accumulate).
// Global->LDS staging uses GLOBAL_LOAD_ASYNC_TO_LDS_B128 (ASYNCcnt) when the
// toolchain exposes the builtin; otherwise falls back to b128 load/ds_store.
// ---------------------------------------------------------------------------

typedef __attribute__((ext_vector_type(16))) __bf16          v16bf;
typedef __attribute__((ext_vector_type(8)))  float           v8f;
typedef __attribute__((ext_vector_type(8)))  unsigned short  ushort8;

union BF16x16 { v16bf v; ushort8 h[2]; };

#define WMMA_BF16(A_, B_, C_) \
  __builtin_amdgcn_wmma_f32_16x16x32_bf16(false, (A_), false, (B_), (short)0, (C_), false, false)

#if defined(__has_builtin)
# if __has_builtin(__builtin_amdgcn_global_load_async_to_lds_b128)
#  define HAVE_ASYNC_LDS 1
# endif
#endif
#ifndef HAVE_ASYNC_LDS
# define HAVE_ASYNC_LDS 0
#endif

#if HAVE_ASYNC_LDS
typedef int vsi4 __attribute__((vector_size(4 * sizeof(int))));
typedef __attribute__((address_space(1))) vsi4* async_gp;   // global v4i*
typedef __attribute__((address_space(3))) vsi4* async_lp;   // LDS v4i*
#endif

__device__ __forceinline__ void async_copy16(const unsigned short* g,
                                             unsigned short* l) {
#if HAVE_ASYNC_LDS
  __builtin_amdgcn_global_load_async_to_lds_b128((async_gp)g, (async_lp)l, 0, 0);
#else
  *(ushort8*)l = *(const ushort8*)g;
#endif
}

__device__ __forceinline__ void wait_async_copies() {
#if HAVE_ASYNC_LDS
# if __has_builtin(__builtin_amdgcn_s_wait_asynccnt)
  __builtin_amdgcn_s_wait_asynccnt(0);
# else
  asm volatile("s_wait_asynccnt 0" ::: "memory");
# endif
#endif
}

__device__ __forceinline__ v8f v8f_zero() {
  v8f z = {0.f, 0.f, 0.f, 0.f, 0.f, 0.f, 0.f, 0.f};
  return z;
}

__device__ __forceinline__ unsigned short f32_to_bf16(float f) {
  unsigned int u = __float_as_uint(f);
  unsigned int r = u + 0x7FFFu + ((u >> 16) & 1u);           // round-to-nearest-even
  if ((u & 0x7F800000u) == 0x7F800000u) r = u;               // keep inf/nan bits
  return (unsigned short)(r >> 16);
}

// ---------------------------------------------------------------------------
// Elementwise f32 -> bf16
// ---------------------------------------------------------------------------
__global__ void cvt_f32_bf16_kernel(const float* __restrict__ in,
                                    unsigned short* __restrict__ out, int n) {
  for (int i = blockIdx.x * blockDim.x + threadIdx.x; i < n;
       i += gridDim.x * blockDim.x)
    out[i] = f32_to_bf16(in[i]);
}

// W is [K][N] row-major f32; produce Wt [N][K] row-major bf16.
__global__ void transpose_cvt_kernel(const float* __restrict__ W,
                                     unsigned short* __restrict__ Wt,
                                     int n, int C_) {
  for (int i = blockIdx.x * blockDim.x + threadIdx.x; i < n;
       i += gridDim.x * blockDim.x) {
    int nrow = i / C_;
    int kcol = i - nrow * C_;
    Wt[i] = f32_to_bf16(W[(size_t)kcol * C_ + nrow]);
  }
}

// ---------------------------------------------------------------------------
// GEMM: out[M][N] = A[M][K](bf16) * Wt[N][K](bf16, pre-transposed) + bias
// mode 0: bf16 scatter to [B,H,T,D] head layout (T=2048,H=16,D=64)
// mode 1: f32 row-major [M][N]
// Block tile 128x128, K-step 32, 256 threads (8 waves), wave w owns rows
// w*16..w*16+15 x all 128 cols (8 wmma accumulators).
// Double-buffered LDS: tile k+1 staged (async) while WMMAs consume tile k.
// ---------------------------------------------------------------------------
#define GT_M 128
#define GT_N 128
#define GT_K 32
#define GPAD 40   // LDS row stride (ushorts); 80B, 16B-aligned

__global__ __launch_bounds__(256)
void gemm_bf16_kernel(const unsigned short* __restrict__ A,
                      const unsigned short* __restrict__ Wt,
                      const float* __restrict__ bias,
                      unsigned short* __restrict__ out_bf16,
                      float* __restrict__ out_f32,
                      int M, int N, int K, int mode) {
  __shared__ __attribute__((aligned(16))) unsigned short Atile[2][GT_M][GPAD];
  __shared__ __attribute__((aligned(16))) unsigned short Btile[2][GT_N][GPAD];

  const int m0   = blockIdx.x * GT_M;
  const int n0   = blockIdx.y * GT_N;
  const int tid  = threadIdx.x;
  const int w    = tid >> 5;        // wave 0..7
  const int lane = tid & 31;
  const int lhalf = lane >> 4;      // 0/1
  const int l16   = lane & 15;

  v8f acc[8];
  for (int i = 0; i < 8; ++i) acc[i] = v8f_zero();

  const int lrow = tid >> 1;              // 0..127
  const int lc0  = (tid & 1) * 16;        // 0 or 16 (ushorts)
  const unsigned short* gA = A  + (size_t)(m0 + lrow) * K + lc0;
  const unsigned short* gB = Wt + (size_t)(n0 + lrow) * K + lc0;

  // stage one 128x32 A-tile + 128x32 B-tile into LDS buffer `buf`
  auto stage = [&](int kofs, int buf) {
    async_copy16(gA + kofs,     &Atile[buf][lrow][lc0]);
    async_copy16(gA + kofs + 8, &Atile[buf][lrow][lc0 + 8]);
    async_copy16(gB + kofs,     &Btile[buf][lrow][lc0]);
    async_copy16(gB + kofs + 8, &Btile[buf][lrow][lc0 + 8]);
  };

  // prologue
  stage(0, 0);
  wait_async_copies();
  __syncthreads();

  int buf = 0;
  for (int k0 = 0; k0 < K; k0 += GT_K) {
    const bool more = (k0 + GT_K) < K;
    if (more) {
      stage(k0 + GT_K, buf ^ 1);                 // overlap with compute
      if (k0 + 2 * GT_K < K) {                   // pull tile k+2 toward L2
        __builtin_prefetch(gA + k0 + 2 * GT_K, 0, 1);
        __builtin_prefetch(gB + k0 + 2 * GT_K, 0, 1);
      }
    }

    // ---- preload all fragments, then issue WMMAs back-to-back ----
    // A fragment: 16x32 bf16, ISA layout (lane=M%16, K-half by lane/16)
    BF16x16 af;
    af.h[0] = *(const ushort8*)&Atile[buf][w * 16 + l16][lhalf * 8];
    af.h[1] = *(const ushort8*)&Atile[buf][w * 16 + l16][16 + lhalf * 8];
    // B fragments: 32x16 bf16; lane=N, K-half by lane/16 (same order as A)
    BF16x16 bf[8];
    for (int ns = 0; ns < 8; ++ns) {
      const unsigned short* bp = &Btile[buf][ns * 16 + l16][lhalf * 8];
      bf[ns].h[0] = *(const ushort8*)bp;
      bf[ns].h[1] = *(const ushort8*)(bp + 16);
    }
    for (int ns = 0; ns < 8; ++ns)
      acc[ns] = WMMA_BF16(af.v, bf[ns].v, acc[ns]);

    if (more) wait_async_copies();   // staged buffer complete before barrier
    __syncthreads();
    buf ^= 1;
  }

  // ---- epilogue: C/D layout  N = lane%16,  M = (lane/16)*8 + r ----
  for (int ns = 0; ns < 8; ++ns) {
    const int n  = n0 + ns * 16 + l16;
    const float bb = bias[n];
    for (int r = 0; r < 8; ++r) {
      const int m  = m0 + w * 16 + lhalf * 8 + r;
      const float val = acc[ns][r] + bb;
      if (mode == 0) {
        // m = b*T + t ; n = h*D + d  ->  [B,H,T,D] bf16
        const int b_ = m >> 11, t_ = m & 2047;
        const int h_ = n >> 6,  d_ = n & 63;
        const size_t idx = ((((size_t)b_ * 16 + h_) << 11) + t_) * 64 + d_;
        out_bf16[idx] = f32_to_bf16(val);
      } else {
        out_f32[(size_t)m * N + n] = val;
      }
    }
  }
}

// ---------------------------------------------------------------------------
// Flash attention over bf16 q/k/v in [B,H,T,D], D=64, T=2048.
// Mask: allowed iff j <= max(i, 1984)  (causal OR context prefix j<1985).
// One block = 128 query rows of one (b,h); 8 waves x 16 rows each.
// Output y as bf16 in [B,T,H*D] (ready as GEMM A-matrix).
// ---------------------------------------------------------------------------
#define T_SEQ 2048
#define DHEAD 64
#define CTX_LIM 1984        // j <= 1984 always allowed
#define APAD 72             // 144B row stride, 16B-aligned

__global__ __launch_bounds__(256)
void flash_attn_kernel(const unsigned short* __restrict__ q,
                       const unsigned short* __restrict__ k,
                       const unsigned short* __restrict__ v,
                       unsigned short* __restrict__ y) {
  __shared__ __attribute__((aligned(16))) unsigned short Qt[128][APAD];
  __shared__ __attribute__((aligned(16))) unsigned short Kt[64][APAD];
  __shared__ __attribute__((aligned(16))) unsigned short Vt[64][APAD];   // Vt[d][j]
  __shared__ __attribute__((aligned(16))) unsigned short Pt[128][APAD];

  const int m0 = blockIdx.x * 128;
  const int bh = blockIdx.y;               // b*H + h
  const int b_ = bh >> 4, h_ = bh & 15;
  const int tid = threadIdx.x;
  const int w = tid >> 5, lane = tid & 31;
  const int lhalf = lane >> 4, l16 = lane & 15;

  const size_t base = (size_t)bh * T_SEQ * DHEAD;

  // ---- load Q tile [128][64] (async -> LDS) ----
  {
    const int row = tid >> 1;
    const int c0  = (tid & 1) * 32;
    const unsigned short* g = q + base + (size_t)(m0 + row) * DHEAD + c0;
    async_copy16(g,      &Qt[row][c0]);
    async_copy16(g + 8,  &Qt[row][c0 + 8]);
    async_copy16(g + 16, &Qt[row][c0 + 16]);
    async_copy16(g + 24, &Qt[row][c0 + 24]);
  }

  v8f o[4];
  for (int i = 0; i < 4; ++i) o[i] = v8f_zero();
  float mrow[8], lsum[8];
  for (int r = 0; r < 8; ++r) { mrow[r] = -__builtin_inff(); lsum[r] = 0.f; }

  const float scale = 0.125f;  // 1/sqrt(64)

  const int kv_row = tid >> 2;           // key row 0..63
  const int kv_c0  = (tid & 3) * 16;     // d offset

  wait_async_copies();
  __syncthreads();

  for (int j0 = 0; j0 < T_SEQ; j0 += 64) {
    // ---- load K tile [j][d] (async) and V tile transposed Vt[d][j] ----
    {
      const unsigned short* gk = k + base + (size_t)(j0 + kv_row) * DHEAD + kv_c0;
      async_copy16(gk,     &Kt[kv_row][kv_c0]);
      async_copy16(gk + 8, &Kt[kv_row][kv_c0 + 8]);
      const ushort8* gv =
          (const ushort8*)(v + base + (size_t)(j0 + kv_row) * DHEAD + kv_c0);
      ushort8 v0 = gv[0], v1 = gv[1];
      for (int e = 0; e < 8; ++e) Vt[kv_c0 + e][kv_row]     = v0[e];
      for (int e = 0; e < 8; ++e) Vt[kv_c0 + 8 + e][kv_row] = v1[e];
    }
    wait_async_copies();
    __syncthreads();

    // ---- S = Q K^T  (16x64 per wave, f32 acc) ----
    v8f s[4];
    for (int i = 0; i < 4; ++i) s[i] = v8f_zero();
    for (int kk = 0; kk < DHEAD; kk += 32) {
      BF16x16 af;
      af.h[0] = *(const ushort8*)&Qt[w * 16 + l16][kk + lhalf * 8];
      af.h[1] = *(const ushort8*)&Qt[w * 16 + l16][kk + 16 + lhalf * 8];
      BF16x16 bfr[4];
      for (int ns = 0; ns < 4; ++ns) {
        const unsigned short* bp = &Kt[ns * 16 + l16][kk + lhalf * 8];
        bfr[ns].h[0] = *(const ushort8*)bp;
        bfr[ns].h[1] = *(const ushort8*)(bp + 16);
      }
      for (int ns = 0; ns < 4; ++ns)
        s[ns] = WMMA_BF16(af.v, bfr[ns].v, s[ns]);
    }

    // ---- mask + online softmax (rows split across wave halves) ----
    float alpha[8];
    for (int r = 0; r < 8; ++r) {
      const int irow = m0 + w * 16 + lhalf * 8 + r;
      const int lim  = irow > CTX_LIM ? irow : CTX_LIM;
      float rmax = -__builtin_inff();
      for (int ns = 0; ns < 4; ++ns) {
        const int jc = j0 + ns * 16 + l16;
        float sv = s[ns][r] * scale;
        sv = (jc <= lim) ? sv : -__builtin_inff();
        s[ns][r] = sv;
        rmax = fmaxf(rmax, sv);
      }
      for (int mk = 1; mk < 16; mk <<= 1)            // stays within wave half
        rmax = fmaxf(rmax, __shfl_xor(rmax, mk, 32));
      const float mnew = fmaxf(mrow[r], rmax);
      alpha[r] = __expf(mrow[r] - mnew);             // first block: exp(-inf)=0
      float rs = 0.f;
      for (int ns = 0; ns < 4; ++ns) {
        const float p = __expf(s[ns][r] - mnew);
        s[ns][r] = p;
        rs += p;
      }
      for (int mk = 1; mk < 16; mk <<= 1)
        rs += __shfl_xor(rs, mk, 32);
      lsum[r] = lsum[r] * alpha[r] + rs;
      mrow[r] = mnew;
    }

    // ---- rescale O; spill P to LDS as bf16 (own-wave strip only) ----
    for (int ns = 0; ns < 4; ++ns)
      for (int r = 0; r < 8; ++r) {
        o[ns][r] *= alpha[r];
        Pt[w * 16 + lhalf * 8 + r][ns * 16 + l16] = f32_to_bf16(s[ns][r]);
      }

    // ---- O += P V  (A=P from LDS, B=V in d-major layout) ----
    for (int kk = 0; kk < 64; kk += 32) {
      BF16x16 af;
      af.h[0] = *(const ushort8*)&Pt[w * 16 + l16][kk + lhalf * 8];
      af.h[1] = *(const ushort8*)&Pt[w * 16 + l16][kk + 16 + lhalf * 8];
      BF16x16 bfr[4];
      for (int ns = 0; ns < 4; ++ns) {
        const unsigned short* bp = &Vt[ns * 16 + l16][kk + lhalf * 8];
        bfr[ns].h[0] = *(const ushort8*)bp;
        bfr[ns].h[1] = *(const ushort8*)(bp + 16);
      }
      for (int ns = 0; ns < 4; ++ns)
        o[ns] = WMMA_BF16(af.v, bfr[ns].v, o[ns]);
    }
    __syncthreads();   // protect Kt/Vt reuse next iteration
  }

  // ---- normalize and emit y[B,T,H*D] bf16 ----
  for (int ns = 0; ns < 4; ++ns)
    for (int r = 0; r < 8; ++r) {
      const int t_ = m0 + w * 16 + lhalf * 8 + r;
      const int d_ = ns * 16 + l16;
      const float val = o[ns][r] / lsum[r];
      y[(size_t)(b_ * T_SEQ + t_) * 1024 + h_ * 64 + d_] = f32_to_bf16(val);
    }
}

// ---------------------------------------------------------------------------
// Host-side orchestration (graph-capture safe: no alloc/sync, stream only).
// ---------------------------------------------------------------------------
extern "C" void kernel_launch(void* const* d_in, const int* in_sizes, int n_in,
                              void* d_out, int out_size, void* d_ws, size_t ws_size,
                              hipStream_t stream) {
  (void)in_sizes; (void)n_in; (void)out_size; (void)ws_size;

  const float* x  = (const float*)d_in[0];
  const float* Wq = (const float*)d_in[1];
  const float* bq = (const float*)d_in[2];
  const float* Wk = (const float*)d_in[3];
  const float* bk = (const float*)d_in[4];
  const float* Wv = (const float*)d_in[5];
  const float* bv = (const float*)d_in[6];
  const float* Wp = (const float*)d_in[7];
  const float* bp = (const float*)d_in[8];
  float* out = (float*)d_out;

  const int B = 4, T = 2048, C = 1024;
  const int MT = B * T;                       // 8192
  const size_t NX = (size_t)MT * C;           // 8,388,608
  const size_t NW = (size_t)C * C;            // 1,048,576

  // workspace carve-out (~92 MB total), 256B aligned
  char* ws = (char*)d_ws;
  size_t off = 0;
  auto carve = [&](size_t bytes) -> void* {
    void* p = ws + off;
    off += (bytes + 255) & ~(size_t)255;
    return p;
  };
  unsigned short* xb  = (unsigned short*)carve(NX * 2);
  unsigned short* Wqt = (unsigned short*)carve(NW * 2);
  unsigned short* Wkt = (unsigned short*)carve(NW * 2);
  unsigned short* Wvt = (unsigned short*)carve(NW * 2);
  unsigned short* Wpt = (unsigned short*)carve(NW * 2);
  unsigned short* qb  = (unsigned short*)carve(NX * 2);   // [B,H,T,D]
  unsigned short* kb  = (unsigned short*)carve(NX * 2);
  unsigned short* vb  = (unsigned short*)carve(NX * 2);
  unsigned short* yb  = (unsigned short*)carve(NX * 2);   // [B,T,C]

  // 1) precision conversion
  cvt_f32_bf16_kernel<<<4096, 256, 0, stream>>>(x, xb, (int)NX);
  transpose_cvt_kernel<<<2048, 256, 0, stream>>>(Wq, Wqt, (int)NW, C);
  transpose_cvt_kernel<<<2048, 256, 0, stream>>>(Wk, Wkt, (int)NW, C);
  transpose_cvt_kernel<<<2048, 256, 0, stream>>>(Wv, Wvt, (int)NW, C);
  transpose_cvt_kernel<<<2048, 256, 0, stream>>>(Wp, Wpt, (int)NW, C);

  // 2) q/k/v projections (WMMA GEMM, scatter to head layout)
  dim3 gg(MT / GT_M, C / GT_N);               // (64, 8)
  gemm_bf16_kernel<<<gg, 256, 0, stream>>>(xb, Wqt, bq, qb, nullptr, MT, C, C, 0);
  gemm_bf16_kernel<<<gg, 256, 0, stream>>>(xb, Wkt, bk, kb, nullptr, MT, C, C, 0);
  gemm_bf16_kernel<<<gg, 256, 0, stream>>>(xb, Wvt, bv, vb, nullptr, MT, C, C, 0);

  // 3) flash attention
  dim3 ga(T / 128, B * 16);                   // (16, 64)
  flash_attn_kernel<<<ga, 256, 0, stream>>>(qb, kb, vb, yb);

  // 4) output projection -> f32 d_out
  gemm_bf16_kernel<<<gg, 256, 0, stream>>>(yb, Wpt, bp, nullptr, out, MT, C, C, 1);
}
